// AttentionBlock_88064009437220
// MI455X (gfx1250) — matrix operand order repaired and checked by reading, not compiled, not run on hardware
//
#include <hip/hip_runtime.h>
#include <stddef.h>

// ---------------- WMMA types ----------------
typedef __attribute__((ext_vector_type(16))) __bf16 bf16x16;
typedef __attribute__((ext_vector_type(8)))  float  f32x8;

struct __align__(16) U128 { unsigned u[4]; };
union Frag { bf16x16 v; U128 q[2]; };

__device__ __forceinline__ f32x8 wmma_bf16(const Frag& a, const Frag& b, f32x8 c) {
  return __builtin_amdgcn_wmma_f32_16x16x32_bf16(false, a.v, false, b.v, (short)0, c,
                                                 false, false);
}

// ---- CDNA5 async global->LDS copy (16B per lane), ASYNCcnt tracked ----
__device__ __forceinline__ void async_ld16(void* lds, const void* gptr) {
  const unsigned l = (unsigned)(unsigned long long)lds;  // low 32b = LDS offset
  asm volatile("global_load_async_to_lds_b128 %0, %1, off"
               :: "v"(l), "v"(gptr) : "memory");
}
__device__ __forceinline__ void wait_async0() {
  asm volatile("s_wait_asynccnt 0" ::: "memory");
}

// ---------------- problem sizes ----------------
constexpr int kB  = 4;
constexpr int kT  = 2048;
constexpr int kE  = 1024;
constexpr int kH  = 16;
constexpr int kDh = 64;
constexpr int kF  = 4096;
constexpr int kN  = kB * kT;   // 8192 token rows

// =====================================================================
// LayerNorm (fp32 in) -> bf16 out.  One block of 256 threads per row.
// =====================================================================
__global__ __launch_bounds__(256)
void ln_bf16_kernel(const float* __restrict__ x, const float* __restrict__ g,
                    const float* __restrict__ bt, __bf16* __restrict__ out) {
  const int row = blockIdx.x;
  const float* xr = x + (size_t)row * kE;
  __shared__ float red[2][8];
  const int t = threadIdx.x;
  float s = 0.f, s2 = 0.f;
  for (int c = t; c < kE; c += 256) { float v = xr[c]; s += v; s2 += v * v; }
  for (int off = 16; off > 0; off >>= 1) {
    s += __shfl_xor(s, off);
    s2 += __shfl_xor(s2, off);
  }
  const int wid = t >> 5, lid = t & 31;
  if (lid == 0) { red[0][wid] = s; red[1][wid] = s2; }
  __syncthreads();
  float sum = 0.f, sumsq = 0.f;
#pragma unroll
  for (int i = 0; i < 8; i++) { sum += red[0][i]; sumsq += red[1][i]; }
  const float mu   = sum / kE;
  const float var  = sumsq / kE - mu * mu;
  const float rstd = rsqrtf(var + 1e-5f);
  for (int c = t; c < kE; c += 256) {
    float v = (xr[c] - mu) * rstd * g[c] + bt[c];
    out[(size_t)row * kE + c] = (__bf16)v;
  }
}

// =====================================================================
// Weight conversions to bf16, stored K-MAJOR ([N][K]) so GEMM B-tiles
// stage into LDS with pure 16B copies (no in-kernel transpose).
// =====================================================================
// Wq/Wk/Wv: [H][E][Dh] fp32 -> Wt[n=h*Dh+d][k=e] bf16
__global__ __launch_bounds__(256)
void cvt_qkv_t_kernel(const float* __restrict__ W, __bf16* __restrict__ out) {
  const int idx = blockIdx.x * 256 + threadIdx.x;  // over H*E*Dh (coalesced read)
  const int d = idx & (kDh - 1);
  const int e = (idx / kDh) & (kE - 1);
  const int h = idx / (kDh * kE);
  out[(size_t)(h * kDh + d) * kE + e] = (__bf16)W[idx];
}

// generic [K][N] fp32 -> [N][K] bf16
__global__ __launch_bounds__(256)
void cvt_t_kernel(const float* __restrict__ in, __bf16* __restrict__ out,
                  int K, int N) {
  const size_t idx = (size_t)blockIdx.x * 256 + threadIdx.x;  // over K*N
  const int k = (int)(idx / N);
  const int n = (int)(idx % N);
  out[(size_t)n * K + k] = (__bf16)in[idx];
}

// =====================================================================
// Generic tiled WMMA GEMM:  C = act(A[M,K] @ Bt[N,K]^T + bias + res)
// 256 threads (8 waves), tile 128x128, K-step 32, double-buffered LDS
// filled with global_load_async_to_lds_b128 (ASYNCcnt) and one barrier
// per K-step.  Each wave: 2x4 WMMA accumulators (32x64 sub-tile).
// OUT_KIND: 0 = fp32 row-major, 1 = bf16 row-major,
//           2 = bf16 head-transposed V layout [B][H][Dh][T]
// =====================================================================
template <int OUT_KIND, bool HAS_BIAS, bool HAS_RES, bool RELU>
__global__ __launch_bounds__(256)
void gemm_kernel(const __bf16* __restrict__ A, const __bf16* __restrict__ Bt,
                 const float* __restrict__ bias, const float* __restrict__ res,
                 void* __restrict__ Cout, int Nn, int K) {
  constexpr int BM = 128, BN = 128, BK = 32;
  constexpr int LD = BK + 8;  // 40 elems = 80B rows (16B aligned)
  __shared__ __align__(16) __bf16 As[2][BM * LD];
  __shared__ __align__(16) __bf16 Bs[2][BN * LD];

  const int t  = threadIdx.x;
  const int m0 = blockIdx.x * BM;
  const int n0 = blockIdx.y * BN;
  const int w    = t >> 5, lane = t & 31;
  const int wm   = (w & 3) * 32;   // wave row offset in tile
  const int wn   = (w >> 2) * 64;  // wave col offset in tile
  const int lnn  = lane & 15, lh = lane >> 4;

  // async-stage one K-step: 2 x 16B for A, 2 x 16B for B per thread
  auto stage = [&](int buf, int k0) {
#pragma unroll
    for (int i = 0; i < 2; i++) {
      const int idx = t + i * 256;       // 0..511
      const int r   = idx >> 2;          // 0..127
      const int c   = (idx & 3) * 8;     // 0..24
      async_ld16(&As[buf][r * LD + c], &A[(size_t)(m0 + r) * K + k0 + c]);
      async_ld16(&Bs[buf][r * LD + c], &Bt[(size_t)(n0 + r) * K + k0 + c]);
    }
  };

  f32x8 acc[2][4];
#pragma unroll
  for (int i = 0; i < 2; i++)
#pragma unroll
    for (int j = 0; j < 4; j++)
#pragma unroll
      for (int r = 0; r < 8; r++) acc[i][j][r] = 0.f;

  const int nSteps = K / BK;
  stage(0, 0);
  for (int s = 0; s < nSteps; s++) {
    const int cur = s & 1;
    wait_async0();       // my wave's stage-s copies landed (in-order)
    __syncthreads();     // everyone's landed; prior readers of buf done
    if (s + 1 < nSteps) stage(cur ^ 1, (s + 1) * BK);  // overlap with compute

    Frag a[2], b[4];
#pragma unroll
    for (int i = 0; i < 2; i++) {
      const __bf16* base = &As[cur][(wm + 16 * i + lnn) * LD];
      a[i].q[0] = *(const U128*)(base + 8 * lh);        // K = 8h..8h+7
      a[i].q[1] = *(const U128*)(base + 16 + 8 * lh);   // K = 16+8h..+7
    }
#pragma unroll
    for (int j = 0; j < 4; j++) {
      const __bf16* base = &Bs[cur][(wn + 16 * j + lnn) * LD];
      b[j].q[0] = *(const U128*)(base + 8 * lh);        // K = 8h..8h+7
      b[j].q[1] = *(const U128*)(base + 16 + 8 * lh);   // K = 16+8h..+7
    }
#pragma unroll
    for (int i = 0; i < 2; i++)
#pragma unroll
      for (int j = 0; j < 4; j++) acc[i][j] = wmma_bf16(a[i], b[j], acc[i][j]);
  }

  // ---- epilogue: C layout lane(l): N=l%16, M=r+8*(l/16) ----
#pragma unroll
  for (int i = 0; i < 2; i++) {
#pragma unroll
    for (int j = 0; j < 4; j++) {
      const int gmb = m0 + wm + 16 * i + 8 * lh;
      const int gn  = n0 + wn + 16 * j + lnn;
      float bv = 0.f;
      if (HAS_BIAS) bv = bias[gn];
#pragma unroll
      for (int r = 0; r < 8; r++) {
        float v = acc[i][j][r] + bv;
        const int gm = gmb + r;
        if (HAS_RES) v += res[(size_t)gm * Nn + gn];
        if (RELU) v = fmaxf(v, 0.f);
        if (OUT_KIND == 0) {
          ((float*)Cout)[(size_t)gm * Nn + gn] = v;
        } else if (OUT_KIND == 1) {
          ((__bf16*)Cout)[(size_t)gm * Nn + gn] = (__bf16)v;
        } else {  // V transposed per head: [B][H][Dh][T]
          const int bb = gm >> 11, tt = gm & (kT - 1);
          const int hh = gn >> 6,  dd = gn & (kDh - 1);
          ((__bf16*)Cout)[(((size_t)(bb * kH + hh) * kDh + dd) << 11) + tt] =
              (__bf16)v;
        }
      }
    }
  }
}

// =====================================================================
// Causal flash attention: one (b,h,64-query-row) tile per 128-thread
// block.  4 waves x 16 query rows; 64-wide key tiles, double-buffered
// async K/V staging; fp32 online softmax; QK^T and PV via WMMA bf16.
// Q/K: bf16 [B*T][E] (head h at cols h*Dh..); V: bf16 [B][H][Dh][T].
// =====================================================================
__global__ __launch_bounds__(128)
void attn_kernel(const __bf16* __restrict__ Q, const __bf16* __restrict__ Kg,
                 const __bf16* __restrict__ Vtr, __bf16* __restrict__ ctx) {
  constexpr int LDK = 72, LDV = 72, LDP = 72;  // 144B rows, 16B aligned
  __shared__ __align__(16) __bf16 Ks[2][64 * LDK];     // [s][d]
  __shared__ __align__(16) __bf16 Vt[2][64 * LDV];     // [d][s]
  __shared__ __align__(16) __bf16 Ps[4 * 16 * LDP];    // per-wave P staging

  const int q0   = blockIdx.x * 64;
  const int h    = blockIdx.y;
  const int b    = blockIdx.z;
  const int hcol = h * kDh;
  const size_t rbase = (size_t)b * kT;
  const size_t vbase = ((size_t)(b * kH + h) * kDh) << 11;  // *kT

  const int t = threadIdx.x, w = t >> 5, lane = t & 31;
  const int lnn = lane & 15, lh = lane >> 4;

  // async-stage one 64-wide key tile (K rows + pre-transposed V rows)
  auto stage = [&](int buf, int s0) {
#pragma unroll
    for (int i = 0; i < 4; i++) {
      const int idx = t + 128 * i;    // 0..511
      const int r   = idx >> 3;       // 0..63
      const int c8  = (idx & 7) * 8;  // 0..56
      async_ld16(&Ks[buf][r * LDK + c8],
                 &Kg[(rbase + s0 + r) * kE + hcol + c8]);
      async_ld16(&Vt[buf][r * LDV + c8],
                 &Vtr[vbase + (size_t)r * kT + s0 + c8]);
    }
  };

  // Q fragments for this wave's 16 rows (A layout), loaded once.
  Frag qf[2];
  {
    const __bf16* qrow = Q + (rbase + q0 + 16 * w + lnn) * kE + hcol;
#pragma unroll
    for (int kk = 0; kk < 2; kk++) {
      qf[kk].q[0] = *(const U128*)(qrow + 32 * kk + 8 * lh);
      qf[kk].q[1] = *(const U128*)(qrow + 32 * kk + 16 + 8 * lh);
    }
  }

  f32x8 o[4];
  float mrow[8], lrow[8];
#pragma unroll
  for (int dt = 0; dt < 4; dt++)
#pragma unroll
    for (int r = 0; r < 8; r++) o[dt][r] = 0.f;
#pragma unroll
  for (int r = 0; r < 8; r++) { mrow[r] = -1e30f; lrow[r] = 0.f; }

  const float scale  = 0.03125f;  // E^-0.5 = 1024^-0.5 (reference scaling)
  const int   nTiles = q0 / 64 + 1;

  stage(0, 0);
  for (int kt = 0; kt < nTiles; kt++) {
    const int cur = kt & 1;
    wait_async0();
    __syncthreads();
    if (kt + 1 < nTiles) stage(cur ^ 1, (kt + 1) * 64);  // overlap

    const int s0 = kt * 64;

    // ---- S = Q K^T  (16x64 per wave, 8 WMMA) ----
    f32x8 sacc[4];
#pragma unroll
    for (int nt = 0; nt < 4; nt++) {
#pragma unroll
      for (int r = 0; r < 8; r++) sacc[nt][r] = 0.f;
#pragma unroll
      for (int kk = 0; kk < 2; kk++) {
        Frag bk;  // B = K^T: lane n = key col, contiguous d along K-dim
        const __bf16* base = &Ks[cur][(16 * nt + lnn) * LDK + 32 * kk];
        bk.q[0] = *(const U128*)(base + 8 * lh);
        bk.q[1] = *(const U128*)(base + 16 + 8 * lh);
        sacc[nt] = wmma_bf16(qf[kk], bk, sacc[nt]);
      }
    }

    // ---- scale + causal mask + online softmax ----
    const bool diag = (s0 == q0);
    float mx[8];
#pragma unroll
    for (int r = 0; r < 8; r++) mx[r] = -1e30f;
#pragma unroll
    for (int nt = 0; nt < 4; nt++) {
#pragma unroll
      for (int r = 0; r < 8; r++) {
        float v = sacc[nt][r] * scale;
        if (diag) {
          const int trow = q0 + 16 * w + r + 8 * lh;
          const int scol = s0 + 16 * nt + lnn;
          if (scol > trow) v = -1e30f;
        }
        sacc[nt][r] = v;
        mx[r] = fmaxf(mx[r], v);
      }
    }
#pragma unroll
    for (int off = 1; off < 16; off <<= 1) {
#pragma unroll
      for (int r = 0; r < 8; r++) mx[r] = fmaxf(mx[r], __shfl_xor(mx[r], off));
    }
    float alpha[8], rs[8];
#pragma unroll
    for (int r = 0; r < 8; r++) {
      const float mn = fmaxf(mrow[r], mx[r]);
      alpha[r] = __expf(mrow[r] - mn);
      mrow[r] = mn;
      rs[r] = 0.f;
    }
#pragma unroll
    for (int nt = 0; nt < 4; nt++)
#pragma unroll
      for (int r = 0; r < 8; r++) {
        const float pv = __expf(sacc[nt][r] - mrow[r]);
        sacc[nt][r] = pv;
        rs[r] += pv;
      }
#pragma unroll
    for (int off = 1; off < 16; off <<= 1) {
#pragma unroll
      for (int r = 0; r < 8; r++) rs[r] += __shfl_xor(rs[r], off);
    }
#pragma unroll
    for (int r = 0; r < 8; r++) lrow[r] = lrow[r] * alpha[r] + rs[r];
#pragma unroll
    for (int dt = 0; dt < 4; dt++)
#pragma unroll
      for (int r = 0; r < 8; r++) o[dt][r] *= alpha[r];

    // ---- stage P (C layout -> LDS row-major bf16) ----
    __bf16* pw = &Ps[w * 16 * LDP];
#pragma unroll
    for (int nt = 0; nt < 4; nt++)
#pragma unroll
      for (int r = 0; r < 8; r++)
        pw[(r + 8 * lh) * LDP + 16 * nt + lnn] = (__bf16)sacc[nt][r];
    __syncthreads();

    // ---- O += P @ V  (8 WMMA) ----
#pragma unroll
    for (int kk = 0; kk < 2; kk++) {
      Frag pa;  // A-layout read-back of P
      const __bf16* pb = &pw[lnn * LDP + 32 * kk];
      pa.q[0] = *(const U128*)(pb + 8 * lh);
      pa.q[1] = *(const U128*)(pb + 16 + 8 * lh);
#pragma unroll
      for (int dt = 0; dt < 4; dt++) {
        Frag vf;  // B = V: lane n = head-dim col, contiguous s along K-dim
        const __bf16* vptr = &Vt[cur][(16 * dt + lnn) * LDV + 32 * kk];
        vf.q[0] = *(const U128*)(vptr + 8 * lh);
        vf.q[1] = *(const U128*)(vptr + 16 + 8 * lh);
        o[dt] = wmma_bf16(pa, vf, o[dt]);
      }
    }
  }

  // ---- epilogue: O / l, write bf16 ctx (head-major concat) ----
#pragma unroll
  for (int dt = 0; dt < 4; dt++) {
#pragma unroll
    for (int r = 0; r < 8; r++) {
      const float v  = o[dt][r] / lrow[r];
      const size_t row = rbase + q0 + 16 * w + r + 8 * lh;
      ctx[row * kE + hcol + 16 * dt + lnn] = (__bf16)v;
    }
  }
}

// =====================================================================
// Host launcher
// =====================================================================
extern "C" void kernel_launch(void* const* d_in, const int* in_sizes, int n_in,
                              void* d_out, int out_size, void* d_ws, size_t ws_size,
                              hipStream_t stream) {
  (void)in_sizes; (void)n_in; (void)out_size; (void)ws_size;
  const float* x   = (const float*)d_in[0];
  const float* Wq  = (const float*)d_in[1];
  const float* Wk  = (const float*)d_in[2];
  const float* Wv  = (const float*)d_in[3];
  const float* Wo  = (const float*)d_in[4];
  const float* bo  = (const float*)d_in[5];
  const float* W1  = (const float*)d_in[6];
  const float* b1  = (const float*)d_in[7];
  const float* W2  = (const float*)d_in[8];
  const float* b2  = (const float*)d_in[9];
  const float* g1  = (const float*)d_in[10];
  const float* bt1 = (const float*)d_in[11];
  const float* g2  = (const float*)d_in[12];
  const float* bt2 = (const float*)d_in[13];

  char* p = (char*)d_ws;
  auto take = [&](size_t bytes) -> char* {
    char* r = p;
    p += (bytes + 255) & ~(size_t)255;
    return r;
  };
  __bf16* hb   = (__bf16*)take((size_t)kN * kE * 2);
  __bf16* Wq2  = (__bf16*)take((size_t)kE * kE * 2);   // [N][K]
  __bf16* Wk2  = (__bf16*)take((size_t)kE * kE * 2);
  __bf16* Wv2  = (__bf16*)take((size_t)kE * kE * 2);
  __bf16* Wo2  = (__bf16*)take((size_t)kE * kE * 2);
  __bf16* W1b  = (__bf16*)take((size_t)kE * kF * 2);
  __bf16* W2b  = (__bf16*)take((size_t)kF * kE * 2);
  __bf16* Qb   = (__bf16*)take((size_t)kN * kE * 2);
  __bf16* Kb   = (__bf16*)take((size_t)kN * kE * 2);
  __bf16* Vtr  = (__bf16*)take((size_t)kN * kE * 2);   // [B][H][Dh][T]
  __bf16* ctxb = (__bf16*)take((size_t)kN * kE * 2);
  float*  x1   = (float*) take((size_t)kN * kE * 4);
  __bf16* h2b  = (__bf16*)take((size_t)kN * kE * 2);
  __bf16* ub   = (__bf16*)take((size_t)kN * kF * 2);

  // 1) LN1 -> bf16
  ln_bf16_kernel<<<kN, 256, 0, stream>>>(x, g1, bt1, hb);

  // 2) weight converts (all K-major)
  cvt_qkv_t_kernel<<<(kH * kE * kDh) / 256, 256, 0, stream>>>(Wq, Wq2);
  cvt_qkv_t_kernel<<<(kH * kE * kDh) / 256, 256, 0, stream>>>(Wk, Wk2);
  cvt_qkv_t_kernel<<<(kH * kE * kDh) / 256, 256, 0, stream>>>(Wv, Wv2);
  cvt_t_kernel<<<(kE * kE) / 256, 256, 0, stream>>>(Wo, Wo2, kE, kE);
  cvt_t_kernel<<<(kE * kF) / 256, 256, 0, stream>>>(W1, W1b, kE, kF);
  cvt_t_kernel<<<(kF * kE) / 256, 256, 0, stream>>>(W2, W2b, kF, kE);

  const dim3 gE(kN / 128, kE / 128);  // (64, 8)
  const dim3 gF(kN / 128, kF / 128);  // (64, 32)

  // 3) QKV projections (V written head-transposed)
  gemm_kernel<1, false, false, false><<<gE, 256, 0, stream>>>(hb, Wq2, nullptr, nullptr, Qb, kE, kE);
  gemm_kernel<1, false, false, false><<<gE, 256, 0, stream>>>(hb, Wk2, nullptr, nullptr, Kb, kE, kE);
  gemm_kernel<2, false, false, false><<<gE, 256, 0, stream>>>(hb, Wv2, nullptr, nullptr, Vtr, kE, kE);

  // 4) causal flash attention
  attn_kernel<<<dim3(kT / 64, kH, kB), 128, 0, stream>>>(Qb, Kb, Vtr, ctxb);

  // 5) out projection + residual:  x1 = x + ctx @ Wo + bo   (fp32)
  gemm_kernel<0, true, true, false><<<gE, 256, 0, stream>>>(ctxb, Wo2, bo, x, x1, kE, kE);

  // 6) LN2 -> bf16
  ln_bf16_kernel<<<kN, 256, 0, stream>>>(x1, g2, bt2, h2b);

  // 7) FFN1: u = relu(h2 @ W1 + b1)  (bf16)
  gemm_kernel<1, true, false, true><<<gF, 256, 0, stream>>>(h2b, W1b, b1, nullptr, ub, kF, kE);

  // 8) FFN2 + residual -> d_out (fp32)
  gemm_kernel<0, true, true, false><<<gE, 256, 0, stream>>>(ub, W2b, b2, x1, (float*)d_out, kE, kF);
}
